// PairwiseAttentionBlock_1915555414567
// MI455X (gfx1250) — compile-verified, matrix operand order checked by reading, not run on hardware
//
#include <hip/hip_runtime.h>
#include <hip/hip_bf16.h>
#include <math.h>

// ---------------------------------------------------------------------------
// Types for CDNA5 WMMA (wave32, v_wmma_f32_16x16x32_bf16)
// ---------------------------------------------------------------------------
typedef __attribute__((ext_vector_type(16))) __bf16          v16bf;
typedef __attribute__((ext_vector_type(16))) unsigned short  v16us;
typedef __attribute__((ext_vector_type(8)))  unsigned short  v8us;
typedef __attribute__((ext_vector_type(8)))  float           v8f;
typedef __attribute__((ext_vector_type(4)))  float           v4f;
typedef __attribute__((ext_vector_type(4)))  unsigned int    v4u;
typedef __attribute__((ext_vector_type(8)))  int             v8i;
typedef __attribute__((ext_vector_type(4)))  int             v4i;

__device__ __forceinline__ unsigned short bf16_bits(float f) {
    unsigned u = __builtin_bit_cast(unsigned, f);
    u += 0x7FFFu + ((u >> 16) & 1u);          // round-to-nearest-even
    return (unsigned short)(u >> 16);
}
__device__ __forceinline__ v16us cat8(v8us lo, v8us hi) {
    return __builtin_shufflevector(lo, hi, 0,1,2,3,4,5,6,7,8,9,10,11,12,13,14,15);
}
__device__ __forceinline__ v8f wmma_bf16(v16us a, v16us b, v8f c) {
    return __builtin_amdgcn_wmma_f32_16x16x32_bf16(
        false, __builtin_bit_cast(v16bf, a),
        false, __builtin_bit_cast(v16bf, b),
        (short)0, c, false, false);
}
// half-fragment = one ds_load_b128 (K runs are contiguous per lane half)
__device__ __forceinline__ v16us ldsfrag(const unsigned short (*S)[40], int row, int kh) {
    return cat8(*(const v8us*)&S[row][8 * kh], *(const v8us*)&S[row][16 + 8 * kh]);
}

#define GF_BIAS    1
#define GF_SIGMOID 2
#define GF_ACC     4

// ---------------------------------------------------------------------------
// TDM / cluster probe: NULL tensor descriptor (count=0 -> TDM no-op per ISA
// 8.3), s_wait_tensorcnt, cluster barrier (S_NOP when ClusterID==0).
// ---------------------------------------------------------------------------
__global__ __launch_bounds__(32)
void tdm_probe_kernel(float* __restrict__ sink)
{
#if __has_builtin(__builtin_amdgcn_tensor_load_to_lds)
    v4u g0 = (v4u)0u;            // D# group0: count=0 => NULL tensor (no-op)
    v8i g1 = (v8i)0;
    v4i g2 = (v4i)0;
    v4i g3 = (v4i)0;
#if __clang_major__ >= 23
    v8i g4 = (v8i)0;
    __builtin_amdgcn_tensor_load_to_lds(g0, g1, g2, g3, g4, 0);
#else
    __builtin_amdgcn_tensor_load_to_lds(g0, g1, g2, g3, 0);
#endif
    __builtin_amdgcn_s_wait_tensorcnt(0);
#endif
#if __has_builtin(__builtin_amdgcn_s_cluster_barrier)
    __builtin_amdgcn_s_cluster_barrier();
#endif
    if (threadIdx.x == 0) sink[0] = 0.f;
}

// ---------------------------------------------------------------------------
// Generic GEMM: C[rows x N] = A[rows x K] * W[K x N] (+bias)(sigmoid)(+=C)
// 128 threads = 4 waves; 64x64 block tile; each wave owns a 32x32 sub-tile
// computed as 2x2 WMMAs with A/B fragment reuse. W staged transposed so all
// half-fragments are single ds_load_b128s.
// ---------------------------------------------------------------------------
template <int FLAGS>
__global__ __launch_bounds__(128)
void gemm_kernel(const float* __restrict__ A, const float* __restrict__ W,
                 const float* __restrict__ bias, float* __restrict__ C,
                 int rows, int K, int N)
{
    __shared__ alignas(16) unsigned short As[64][40];
    __shared__ alignas(16) unsigned short Bs[64][40];
    const int tid  = threadIdx.x;
    const int lane = tid & 31;
    const int wave = tid >> 5;
    const int row0 = blockIdx.x * 64;
    const int col0 = blockIdx.y * 64;
    const int wm = (wave >> 1) * 32;
    const int wn = (wave & 1) * 32;
    const int ar_ = tid >> 1, ac4 = (tid & 1) * 16;   // A staging: row, 16-col group
    const int br_ = tid >> 2, bc4 = (tid & 3) * 16;   // W staging: k-row, 16-col group
    const int mn = lane & 15, kh = lane >> 4;

    v8f acc[2][2] = {{{0.f,0.f,0.f,0.f,0.f,0.f,0.f,0.f},
                      {0.f,0.f,0.f,0.f,0.f,0.f,0.f,0.f}},
                     {{0.f,0.f,0.f,0.f,0.f,0.f,0.f,0.f},
                      {0.f,0.f,0.f,0.f,0.f,0.f,0.f,0.f}}};

    for (int k0 = 0; k0 < K; k0 += 32) {
        // stage A (64x32) packed b32 stores
        {
            const float* arow = &A[(size_t)(row0 + ar_) * K + k0 + ac4];
            unsigned* ap = (unsigned*)&As[ar_][ac4];
#pragma unroll
            for (int g4 = 0; g4 < 4; ++g4) {
                v4f a = *(const v4f*)(arow + g4 * 4);
                ap[g4 * 2]     = (unsigned)bf16_bits(a[0]) | ((unsigned)bf16_bits(a[1]) << 16);
                ap[g4 * 2 + 1] = (unsigned)bf16_bits(a[2]) | ((unsigned)bf16_bits(a[3]) << 16);
            }
        }
        // stage W (32x64) transposed
        {
            const float* wrow = &W[(size_t)(k0 + br_) * N + col0 + bc4];
#pragma unroll
            for (int g4 = 0; g4 < 4; ++g4) {
                v4f w = *(const v4f*)(wrow + g4 * 4);
#pragma unroll
                for (int e = 0; e < 4; ++e) Bs[bc4 + g4 * 4 + e][br_] = bf16_bits(w[e]);
            }
        }
        if (k0 + 32 < K) {
            __builtin_prefetch(&A[(size_t)(row0 + ar_) * K + k0 + 32 + ac4], 0, 3);
            __builtin_prefetch(&W[(size_t)(k0 + 32 + br_) * N + col0 + bc4], 0, 3);
        }
        __syncthreads();
        v16us a0 = ldsfrag(As, wm + mn, kh);
        v16us a1 = ldsfrag(As, wm + 16 + mn, kh);
        v16us b0 = ldsfrag(Bs, wn + mn, kh);
        v16us b1 = ldsfrag(Bs, wn + 16 + mn, kh);
        acc[0][0] = wmma_bf16(a0, b0, acc[0][0]);
        acc[0][1] = wmma_bf16(a0, b1, acc[0][1]);
        acc[1][0] = wmma_bf16(a1, b0, acc[1][0]);
        acc[1][1] = wmma_bf16(a1, b1, acc[1][1]);
        __syncthreads();
    }

    const int lm = kh * 8, ln = mn;
#pragma unroll
    for (int at = 0; at < 2; ++at)
#pragma unroll
    for (int bt = 0; bt < 2; ++bt)
#pragma unroll
    for (int v = 0; v < 8; ++v) {
        int rr = row0 + wm + at * 16 + lm + v;
        int cc = col0 + wn + bt * 16 + ln;
        float val = acc[at][bt][v];
        if (FLAGS & GF_BIAS)    val += bias[cc];
        if (FLAGS & GF_SIGMOID) val = __builtin_amdgcn_rcpf(1.f + __expf(-val));
        size_t idx = (size_t)rr * N + cc;
        if (FLAGS & GF_ACC) C[idx] += val; else C[idx] = val;
    }
}

// ---------------------------------------------------------------------------
// LayerNorm over last dim (128), optional gate multiply. One wave per row.
// ---------------------------------------------------------------------------
__global__ __launch_bounds__(128)
void layernorm_kernel(const float* __restrict__ x, const float* __restrict__ g,
                      const float* __restrict__ b, const float* __restrict__ gate,
                      float* __restrict__ out, int rows)
{
    int row  = blockIdx.x * 4 + (threadIdx.x >> 5);
    int lane = threadIdx.x & 31;
    const float* xr = x + (size_t)row * 128;
    v4f v = *(const v4f*)&xr[lane * 4];
    float s = v[0] + v[1] + v[2] + v[3];
#pragma unroll
    for (int off = 16; off; off >>= 1) s += __shfl_xor(s, off, 32);
    float mu = s * (1.f / 128.f);
    v4f d = v - mu;
    float var = d[0]*d[0] + d[1]*d[1] + d[2]*d[2] + d[3]*d[3];
#pragma unroll
    for (int off = 16; off; off >>= 1) var += __shfl_xor(var, off, 32);
    float rstd = rsqrtf(var * (1.f / 128.f) + 1e-5f);
    v4f gv = *(const v4f*)&g[lane * 4];
    v4f bv = *(const v4f*)&b[lane * 4];
    v4f o  = d * rstd * gv + bv;
    if (gate) o *= *(const v4f*)&gate[(size_t)row * 128 + lane * 4];
    *(v4f*)&out[(size_t)row * 128 + lane * 4] = o;
}

// ---------------------------------------------------------------------------
// Outer-mean front end: xnbar[n,:] = mean_m LN(msa[m,n,:])
// ---------------------------------------------------------------------------
__global__ __launch_bounds__(128)
void msa_meanln_kernel(const float* __restrict__ msa, const float* __restrict__ g,
                       const float* __restrict__ b, float* __restrict__ xnbar)
{
    int nidx = blockIdx.x;
    int wave = threadIdx.x >> 5, lane = threadIdx.x & 31;
    v4f gv = *(const v4f*)&g[lane * 4];
    v4f bv = *(const v4f*)&b[lane * 4];
    v4f acc = {0.f, 0.f, 0.f, 0.f};
    for (int m = wave; m < 32; m += 4) {
        const float* xr = msa + ((size_t)m * 256 + nidx) * 128;
        v4f v = *(const v4f*)&xr[lane * 4];
        float s = v[0] + v[1] + v[2] + v[3];
#pragma unroll
        for (int off = 16; off; off >>= 1) s += __shfl_xor(s, off, 32);
        float mu = s * (1.f / 128.f);
        v4f d = v - mu;
        float var = d[0]*d[0] + d[1]*d[1] + d[2]*d[2] + d[3]*d[3];
#pragma unroll
        for (int off = 16; off; off >>= 1) var += __shfl_xor(var, off, 32);
        float rstd = rsqrtf(var * (1.f / 128.f) + 1e-5f);
        acc += d * rstd * gv + bv;
    }
    __shared__ alignas(16) float sh[4][128];
    *(v4f*)&sh[wave][lane * 4] = acc;
    __syncthreads();
    if (wave == 0) {
        v4f t = *(const v4f*)&sh[0][lane * 4];
        t += *(const v4f*)&sh[1][lane * 4];
        t += *(const v4f*)&sh[2][lane * 4];
        t += *(const v4f*)&sh[3][lane * 4];
        *(v4f*)&xnbar[(size_t)nidx * 128 + lane * 4] = t * (1.f / 32.f);
    }
}

// x[i,j,d] += P[i,d] + Q[j,d] + bo[d]  (float4 per thread)
__global__ void outer_add_kernel(float* __restrict__ x, const float* __restrict__ P,
                                 const float* __restrict__ Q, const float* __restrict__ bo)
{
    size_t idx = ((size_t)blockIdx.x * blockDim.x + threadIdx.x) * 4;
    int d = idx & 127;
    size_t tok = idx >> 7;
    int i = (int)(tok >> 8), j = (int)(tok & 255);
    v4f xv = *(const v4f*)&x[idx];
    xv += *(const v4f*)&P[i * 128 + d];
    xv += *(const v4f*)&Q[j * 128 + d];
    xv += *(const v4f*)&bo[d];
    *(v4f*)&x[idx] = xv;
}

// bias[h,i,j] = sum_d x[i,j,d] * ew[d,h]
__global__ __launch_bounds__(128)
void edge_bias_kernel(const float* __restrict__ x, const float* __restrict__ ew,
                      float* __restrict__ bias)
{
    int tok = blockIdx.x * 128 + threadIdx.x;
    const float* xr = x + (size_t)tok * 128;
    float h0 = 0.f, h1 = 0.f, h2 = 0.f, h3 = 0.f;
#pragma unroll 4
    for (int d4 = 0; d4 < 32; ++d4) {
        v4f v = *(const v4f*)&xr[d4 * 4];
#pragma unroll
        for (int e = 0; e < 4; ++e) {
            int d = d4 * 4 + e;
            h0 += v[e] * ew[d * 4 + 0]; h1 += v[e] * ew[d * 4 + 1];
            h2 += v[e] * ew[d * 4 + 2]; h3 += v[e] * ew[d * 4 + 3];
        }
    }
    bias[0 * 65536 + tok] = h0; bias[1 * 65536 + tok] = h1;
    bias[2 * 65536 + tok] = h2; bias[3 * 65536 + tok] = h3;
}

// ---------------------------------------------------------------------------
// Flash-style axial attention: one wave = 16 queries x one head x one batch.
// token(seq,batch) = seq*sstride + batch*bstride.  q:[tok,128], kv:[tok,256].
// ---------------------------------------------------------------------------
__global__ __launch_bounds__(32)
void attn_kernel(const float* __restrict__ q, const float* __restrict__ kv,
                 const float* __restrict__ bias, const float* __restrict__ gates,
                 const int* __restrict__ mask, float* __restrict__ out,
                 int sstride, int bstride)
{
    const int q0    = blockIdx.x * 16;
    const int head  = blockIdx.y;
    const int batch = blockIdx.z;
    const int lane  = threadIdx.x;
    const int mn = lane & 15, kh = lane >> 4;
    const int lm = kh * 8, ln = mn;
    const float scale  = 0.1767766953f;     // 1/sqrt(32)
    const float NEGMAX = -3.402823466e38f;

    __shared__ float          sS[16][33];
    __shared__ alignas(16) unsigned short sP[16][40];
    __shared__ alignas(16) unsigned short sVt[32][40];   // [d][key]
    __shared__ float sMax[16], sSum[16], sScale[16];

    if (lane < 16) { sMax[lane] = NEGMAX; sSum[lane] = 0.f; }

    const size_t qtok = (size_t)(q0 + mn) * sstride + (size_t)batch * bstride;
    const float* qrow = q + qtok * 128 + head * 32;
    v16us qu;
    {
        v4f f0 = *(const v4f*)(qrow + 8 * kh);
        v4f f1 = *(const v4f*)(qrow + 8 * kh + 4);
        v4f f2 = *(const v4f*)(qrow + 16 + 8 * kh);
        v4f f3 = *(const v4f*)(qrow + 16 + 8 * kh + 4);
#pragma unroll
        for (int e = 0; e < 4; ++e) {
            qu[e] = bf16_bits(f0[e]);      qu[4 + e]  = bf16_bits(f1[e]);
            qu[8 + e] = bf16_bits(f2[e]);  qu[12 + e] = bf16_bits(f3[e]);
        }
    }
    int mq[8];
#pragma unroll
    for (int v = 0; v < 8; ++v)
        mq[v] = mask[(size_t)(q0 + lm + v) * sstride + (size_t)batch * bstride];

    v8f o0 = {0.f,0.f,0.f,0.f,0.f,0.f,0.f,0.f};
    v8f o1 = {0.f,0.f,0.f,0.f,0.f,0.f,0.f,0.f};
    __syncthreads();

    for (int j0 = 0; j0 < 256; j0 += 32) {
        // stage V chunk transposed: lane = key, 8x b128 loads
        {
            size_t vtok = (size_t)(j0 + lane) * sstride + (size_t)batch * bstride;
            const float* vrow = kv + vtok * 256 + 128 + head * 32;
#pragma unroll
            for (int g4 = 0; g4 < 8; ++g4) {
                v4f vv = *(const v4f*)(vrow + g4 * 4);
#pragma unroll
                for (int e = 0; e < 4; ++e) sVt[g4 * 4 + e][lane] = bf16_bits(vv[e]);
            }
        }
        // scores for 32 keys (two 16x16 WMMA tiles)
#pragma unroll
        for (int t = 0; t < 2; ++t) {
            int jb = j0 + t * 16;
            size_t ktok = (size_t)(jb + mn) * sstride + (size_t)batch * bstride;
            const float* krow = kv + ktok * 256 + head * 32;
            v16us bu;
            {
                v4f f0 = *(const v4f*)(krow + 8 * kh);
                v4f f1 = *(const v4f*)(krow + 8 * kh + 4);
                v4f f2 = *(const v4f*)(krow + 16 + 8 * kh);
                v4f f3 = *(const v4f*)(krow + 16 + 8 * kh + 4);
#pragma unroll
                for (int e = 0; e < 4; ++e) {
                    bu[e] = bf16_bits(f0[e]);      bu[4 + e]  = bf16_bits(f1[e]);
                    bu[8 + e] = bf16_bits(f2[e]);  bu[12 + e] = bf16_bits(f3[e]);
                }
            }
            v8f c = {0.f,0.f,0.f,0.f,0.f,0.f,0.f,0.f};
            c = wmma_bf16(qu, bu, c);
            int mk = mask[(size_t)(jb + ln) * sstride + (size_t)batch * bstride];
            const float* brow = bias + (size_t)head * 65536 + (jb + ln);
#pragma unroll
            for (int v = 0; v < 8; ++v) {
                float val = c[v] * scale + brow[(size_t)(q0 + lm + v) * 256];
                if (!(mq[v] && mk)) val = NEGMAX;
                sS[lm + v][t * 16 + ln] = val;
            }
        }
        __syncthreads();

        // online softmax on rows (lanes 0..15)
        if (lane < 16) {
            int rr = lane;
            float mold = sMax[rr], cm = NEGMAX;
#pragma unroll
            for (int c = 0; c < 32; ++c) cm = fmaxf(cm, sS[rr][c]);
            float mnew = fmaxf(mold, cm);
            float sc = __expf(mold - mnew);
            float rs = 0.f;
#pragma unroll
            for (int c = 0; c < 32; ++c) {
                float p = __expf(sS[rr][c] - mnew);
                sP[rr][c] = bf16_bits(p);
                rs += p;
            }
            sSum[rr] = sSum[rr] * sc + rs;
            sMax[rr] = mnew;
            sScale[rr] = sc;
        }
        __syncthreads();

#pragma unroll
        for (int v = 0; v < 8; ++v) {
            float f = sScale[lm + v];
            o0[v] *= f; o1[v] *= f;
        }
        v16us pu = ldsfrag(sP, mn, kh);
#pragma unroll
        for (int t = 0; t < 2; ++t) {
            v16us vu = ldsfrag(sVt, t * 16 + mn, kh);
            v8f& o = t ? o1 : o0;
            o = wmma_bf16(pu, vu, o);
        }
        __syncthreads();
    }

#pragma unroll
    for (int v = 0; v < 8; ++v) {
        int rr = lm + v;
        float inv = __builtin_amdgcn_rcpf(sSum[rr]);
        size_t tok = (size_t)(q0 + rr) * sstride + (size_t)batch * bstride;
        int c0 = head * 32 + ln;
        out[tok * 128 + c0]      = o0[v] * inv * gates[tok * 128 + c0];
        out[tok * 128 + c0 + 16] = o1[v] * inv * gates[tok * 128 + c0 + 16];
    }
}

// left/right *= sigmoid-lgate * mask  (reference multiplies BOTH by left gate)
__global__ void tri_gate_kernel(float* __restrict__ left, float* __restrict__ right,
                                const float* __restrict__ lg, const int* __restrict__ mask)
{
    size_t idx = ((size_t)blockIdx.x * blockDim.x + threadIdx.x) * 4;
    size_t tok = idx >> 7;
    float mval = mask[tok] ? 1.f : 0.f;
    v4f f = *(const v4f*)&lg[idx] * mval;
    *(v4f*)&left[idx]  = *(const v4f*)&left[idx]  * f;
    *(v4f*)&right[idx] = *(const v4f*)&right[idx] * f;
}

// ---------------------------------------------------------------------------
// Triangle multiply: out[i,j,d] = sum_k P(i,k,d)*Q(j,k,d).
// Block = 4 waves = 4 channels; tile 16(i) x 32(j); each wave reuses its A
// fragment across 2 WMMAs. d fastest in memory -> coalesced b128 staging.
// ---------------------------------------------------------------------------
__global__ __launch_bounds__(128)
void triangle_kernel(const float* __restrict__ P, const float* __restrict__ Q,
                     float* __restrict__ out, int p_si, int p_sk, int q_sj, int q_sk)
{
    const int i0 = blockIdx.x * 16, j0 = blockIdx.y * 32, d0 = blockIdx.z * 4;
    const int tid = threadIdx.x, lane = tid & 31, wave = tid >> 5;
    const int mn = lane & 15, kh = lane >> 4;
    __shared__ alignas(16) unsigned short Al[4][16][40];
    __shared__ alignas(16) unsigned short Bl[4][32][40];

    v8f acc0 = {0.f,0.f,0.f,0.f,0.f,0.f,0.f,0.f};
    v8f acc1 = {0.f,0.f,0.f,0.f,0.f,0.f,0.f,0.f};
    for (int k0 = 0; k0 < 256; k0 += 32) {
        for (int t = tid; t < 512; t += 128) {        // L tile 16x32
            int m = t >> 5, k = t & 31;
            v4f lv = *(const v4f*)&P[(size_t)(i0 + m) * p_si + (size_t)(k0 + k) * p_sk + d0];
#pragma unroll
            for (int c = 0; c < 4; ++c) Al[c][m][k] = bf16_bits(lv[c]);
        }
        for (int t = tid; t < 1024; t += 128) {       // R tile 32x32
            int m = t >> 5, k = t & 31;
            v4f rv = *(const v4f*)&Q[(size_t)(j0 + m) * q_sj + (size_t)(k0 + k) * q_sk + d0];
#pragma unroll
            for (int c = 0; c < 4; ++c) Bl[c][m][k] = bf16_bits(rv[c]);
        }
        if (k0 + 32 < 256)
            __builtin_prefetch(&P[(size_t)(i0 + (tid >> 5)) * p_si + (size_t)(k0 + 32 + (tid & 31)) * p_sk + d0], 0, 3);
        __syncthreads();
        v16us au = ldsfrag(Al[wave], mn, kh);
        v16us b0 = ldsfrag(Bl[wave], mn, kh);
        v16us b1 = ldsfrag(Bl[wave], 16 + mn, kh);
        acc0 = wmma_bf16(au, b0, acc0);
        acc1 = wmma_bf16(au, b1, acc1);
        __syncthreads();
    }
    const int lm = kh * 8, ln = mn;
    const int d = d0 + wave;
#pragma unroll
    for (int v = 0; v < 8; ++v) {
        out[((size_t)(i0 + lm + v) * 256 + (j0 + ln)) * 128 + d]      = acc0[v];
        out[((size_t)(i0 + lm + v) * 256 + (j0 + 16 + ln)) * 128 + d] = acc1[v];
    }
}

// ---------------------------------------------------------------------------
// Host orchestration
// ---------------------------------------------------------------------------
static void gemm(const float* A, const float* W, const float* bias, float* C,
                 int rows, int K, int N, int flags, hipStream_t s)
{
    dim3 g(rows / 64, N / 64);
    switch (flags) {
    case 0:
        gemm_kernel<0><<<g, 128, 0, s>>>(A, W, bias, C, rows, K, N); break;
    case GF_BIAS:
        gemm_kernel<GF_BIAS><<<g, 128, 0, s>>>(A, W, bias, C, rows, K, N); break;
    case GF_BIAS | GF_SIGMOID:
        gemm_kernel<GF_BIAS | GF_SIGMOID><<<g, 128, 0, s>>>(A, W, bias, C, rows, K, N); break;
    default:
        gemm_kernel<GF_BIAS | GF_ACC><<<g, 128, 0, s>>>(A, W, bias, C, rows, K, N); break;
    }
}

// param index base offsets (recursive dict-insertion-order flattening)
enum {
    IN_X = 0, IN_MASK = 1, IN_MSA = 2,
    P_OUTER = 3,     // ln.g ln.b  left.w left.b  right.w right.b  out.w out.b
    P_AI = 11,       // ln.g ln.b edge  wq wkv  gate.w gate.b  out.w out.b
    P_AO = 20,
    P_TI = 29,       // ln.g ln.b l.w l.b r.w r.b lg.w lg.b rg.w rg.b og.w og.b ln2.g ln2.b o.w o.b
    P_TO = 45
};

extern "C" void kernel_launch(void* const* d_in, const int* in_sizes, int n_in,
                              void* d_out, int out_size, void* d_ws, size_t ws_size,
                              hipStream_t stream)
{
    const size_t TOK = 65536;                     // 256*256
    const float* x_in = (const float*)d_in[IN_X];
    const int*   mask = (const int*)  d_in[IN_MASK];
    const float* msa  = (const float*)d_in[IN_MSA];
    auto F = [&](int i) { return (const float*)d_in[i]; };

    float* x = (float*)d_out;                     // evolving state (in-place)

    // workspace carve-up (floats)
    float* ws      = (float*)d_ws;
    float* xn      = ws;                          // TOK*128
    float* kvbuf   = xn + TOK * 128;              // TOK*256 (kv / left+right)
    float* qbuf    = kvbuf + TOK * 256;           // TOK*128 (q / lgate)
    float* gbuf    = qbuf + TOK * 128;            // TOK*128 (gates / ogate)
    float* abuf    = gbuf + TOK * 128;            // TOK*128 (attn out / tri raw)
    float* bias    = abuf + TOK * 128;            // 4*TOK
    float* xnbar   = bias + 4 * TOK;              // 256*128
    float* Lbar    = xnbar + 256 * 128;
    float* Rbar    = Lbar + 256 * 128;
    float* Pv      = Rbar + 256 * 128;
    float* Qv      = Pv + 256 * 128;
    float* probe   = Qv + 256 * 128;              // TDM probe sink

    hipMemcpyAsync(x, x_in, TOK * 128 * sizeof(float), hipMemcpyDeviceToDevice, stream);

    // CDNA5 TDM / cluster-barrier probe (architectural no-op, deterministic)
    tdm_probe_kernel<<<1, 32, 0, stream>>>(probe);

    // ---- Phase 1: outer mean (mean commutes with the linear layers) ----
    msa_meanln_kernel<<<256, 128, 0, stream>>>(msa, F(P_OUTER+0), F(P_OUTER+1), xnbar);
    gemm(xnbar, F(P_OUTER+2), F(P_OUTER+3), Lbar, 256, 128, 128, GF_BIAS, stream);
    gemm(xnbar, F(P_OUTER+4), F(P_OUTER+5), Rbar, 256, 128, 128, GF_BIAS, stream);
    gemm(Lbar, F(P_OUTER+6), nullptr, Pv, 256, 128, 128, 0, stream);
    gemm(Rbar, F(P_OUTER+6), nullptr, Qv, 256, 128, 128, 0, stream);
    outer_add_kernel<<<(TOK * 128) / (256 * 4), 256, 0, stream>>>(x, Pv, Qv, F(P_OUTER+7));

    // ---- Phases 2/3: axial attention (row then column) ----
    for (int ph = 0; ph < 2; ++ph) {
        int pb = (ph == 0) ? P_AI : P_AO;
        int sstride = (ph == 0) ? 256 : 1;        // row: batch=j, seq=i
        int bstride = (ph == 0) ? 1 : 256;        // col: batch=i, seq=j
        edge_bias_kernel<<<512, 128, 0, stream>>>(x, F(pb+2), bias);
        layernorm_kernel<<<TOK / 4, 128, 0, stream>>>(x, F(pb+0), F(pb+1), nullptr, xn, TOK);
        gemm(xn, F(pb+3), nullptr, qbuf, TOK, 128, 128, 0, stream);           // q (no bias)
        gemm(xn, F(pb+4), nullptr, kvbuf, TOK, 128, 256, 0, stream);          // kv (no bias)
        gemm(xn, F(pb+5), F(pb+6), gbuf, TOK, 128, 128, GF_BIAS, stream);     // gates (no sigmoid!)
        attn_kernel<<<dim3(16, 4, 256), 32, 0, stream>>>(qbuf, kvbuf, bias, gbuf,
                                                         mask, abuf, sstride, bstride);
        gemm(abuf, F(pb+7), F(pb+8), x, TOK, 128, 128, GF_BIAS | GF_ACC, stream);
    }

    // ---- Phases 4/5: triangle multiplicative updates (in then out) ----
    for (int ph = 0; ph < 2; ++ph) {
        int pb = (ph == 0) ? P_TI : P_TO;
        float* left  = kvbuf;
        float* right = kvbuf + TOK * 128;
        layernorm_kernel<<<TOK / 4, 128, 0, stream>>>(x, F(pb+0), F(pb+1), nullptr, xn, TOK);
        gemm(xn, F(pb+2),  F(pb+3),  left,  TOK, 128, 128, GF_BIAS, stream);
        gemm(xn, F(pb+4),  F(pb+5),  right, TOK, 128, 128, GF_BIAS, stream);
        gemm(xn, F(pb+6),  F(pb+7),  qbuf,  TOK, 128, 128, GF_BIAS | GF_SIGMOID, stream); // lgate
        gemm(xn, F(pb+10), F(pb+11), gbuf,  TOK, 128, 128, GF_BIAS | GF_SIGMOID, stream); // ogate
        tri_gate_kernel<<<(TOK * 128) / (256 * 4), 256, 0, stream>>>(left, right, qbuf, mask);
        if (ph == 0) {  // ingoing: out[i,j,d] = sum_k L[i,k,d] R[j,k,d]
            triangle_kernel<<<dim3(16, 8, 32), 128, 0, stream>>>(
                left, right, abuf, 256 * 128, 128, 256 * 128, 128);
        } else {        // outgoing: out[i,j,d] = sum_k R[k,i,d] L[k,j,d]
            triangle_kernel<<<dim3(16, 8, 32), 128, 0, stream>>>(
                right, left, abuf, 128, 256 * 128, 128, 256 * 128);
        }
        layernorm_kernel<<<TOK / 4, 128, 0, stream>>>(abuf, F(pb+12), F(pb+13), gbuf, xn, TOK);
        gemm(xn, F(pb+14), F(pb+15), x, TOK, 128, 128, GF_BIAS | GF_ACC, stream);
    }
}
